// FeatureAggregation_12841952215389
// MI455X (gfx1250) — matrix operand order repaired and checked by reading, not compiled, not run on hardware
//
#include <hip/hip_runtime.h>
#include <hip/hip_bf16.h>

typedef __attribute__((ext_vector_type(16))) _Float16 v16h;
typedef __attribute__((ext_vector_type(8)))  _Float16 v8h;
typedef __attribute__((ext_vector_type(8)))  float    v8f;

#define BATCH 8
#define HW    4096
#define CH    64

// ---------------------------------------------------------------------------
// Kernel 1: fused 1x1-conv projections.  q,k -> f16 [b][n][32] row-major
// (k stored transposed so its rows are kT[m][d]), v -> f16 [b][c][m].
// Weights staged in LDS (broadcast reads, conflict-free).
// ---------------------------------------------------------------------------
__global__ __launch_bounds__(256) void proj_kernel(
    const float* __restrict__ x1, const float* __restrict__ x2,
    const float* __restrict__ Wq, const float* __restrict__ bq,
    const float* __restrict__ Wk, const float* __restrict__ bk,
    const float* __restrict__ Wv,
    _Float16* __restrict__ Q, _Float16* __restrict__ K, _Float16* __restrict__ V)
{
    __shared__ float sWq[32*64], sWk[32*64], sWv[64*64], sbq[32], sbk[32];
    for (int i = threadIdx.x; i < 32*64; i += 256) { sWq[i] = Wq[i]; sWk[i] = Wk[i]; }
    for (int i = threadIdx.x; i < 64*64; i += 256) sWv[i] = Wv[i];
    if (threadIdx.x < 32) { sbq[threadIdx.x] = bq[threadIdx.x]; sbk[threadIdx.x] = bk[threadIdx.x]; }
    __syncthreads();

    int t = blockIdx.x * 256 + threadIdx.x;
    int b = t >> 12;
    int n = t & (HW - 1);

    float xr[64];
#pragma unroll
    for (int c = 0; c < 32; ++c) xr[c]      = x1[(size_t)(b*32 + c)*HW + n];
#pragma unroll
    for (int c = 0; c < 32; ++c) xr[32 + c] = x2[(size_t)(b*32 + c)*HW + n];

    for (int d = 0; d < 32; ++d) {
        float aq = sbq[d], ak = sbk[d];
#pragma unroll
        for (int c = 0; c < 64; ++c) { aq += sWq[d*64 + c] * xr[c]; ak += sWk[d*64 + c] * xr[c]; }
        Q[((size_t)b*HW + n)*32 + d] = (_Float16)aq;
        K[((size_t)b*HW + n)*32 + d] = (_Float16)ak;
    }
    for (int o = 0; o < 64; ++o) {
        float av = 0.f;
#pragma unroll
        for (int c = 0; c < 64; ++c) av += sWv[o*64 + c] * xr[c];
        V[((size_t)b*64 + o)*HW + n] = (_Float16)av;
    }
}

// ---------------------------------------------------------------------------
// Kernel 2: column softmax normalizers.  Zinv[b][m] = 1 / sum_n exp(S[n,m]).
// One wave per 16-column m-tile; loops all 256 n-tiles, one WMMA each.
// C/D layout: lanes L and L+16 hold the same column (rows 0-7 / 8-15), so a
// per-lane f32 accumulate + xor-16 shuffle yields the column sum.
// ---------------------------------------------------------------------------
__global__ __launch_bounds__(128) void colstats_kernel(
    const _Float16* __restrict__ Q, const _Float16* __restrict__ K,
    float* __restrict__ Zinv)
{
    int lane = threadIdx.x & 31;
    int wv   = threadIdx.x >> 5;
    int w    = blockIdx.x * 4 + wv;
    int b    = w >> 8;                 // 256 m-tiles per batch
    int m0   = (w & 255) << 4;
    int hl   = lane & 15, sel = lane >> 4;

    // B operand (kT): lane<16 -> col m0+hl, K(d)=0..15 ; lane>=16 -> K=16..31
    v16h bk = *(const v16h*)(K + ((size_t)b*HW + m0 + hl)*32 + sel*16);

    v8f acc = {};
    const _Float16* qb = Q + (size_t)b*HW*32;
    for (int n0 = 0; n0 < HW; n0 += 16) {
        // A operand (q): lane<16 -> row n0+hl, K = 0..15 ; lane>=16 -> 16..31
        const _Float16* qr = qb + (size_t)(n0 + hl)*32 + sel*16;
        v8h lo = *(const v8h*)(qr);
        v8h hi = *(const v8h*)(qr + 8);
        v16h a;
#pragma unroll
        for (int i = 0; i < 8; ++i) { a[i] = lo[i]; a[8 + i] = hi[i]; }
        v8f s = {};
        s = __builtin_amdgcn_wmma_f32_16x16x32_f16(false, a, false, bk, (short)0, s, false, false);
#pragma unroll
        for (int r = 0; r < 8; ++r) acc[r] += __expf(s[r]);
    }
    float tot = 0.f;
#pragma unroll
    for (int r = 0; r < 8; ++r) tot += acc[r];
    tot += __shfl_xor(tot, 16, 32);
    if (lane < 16) Zinv[(size_t)b*HW + m0 + lane] = 1.0f / tot;
}

// ---------------------------------------------------------------------------
// Kernel 3: attention + residual + final Wsc projection, fully fused.
// Wave owns n-tile [n0,n0+16) x all 64 channels.  Per 32-wide m-step:
//   2 WMMA (S tiles) -> exp * Zinv -> f16 P staged through LDS (layout fix)
//   -> 4 WMMA accumulating OUT[16n x 64c].
// The 4 waves per block are fully independent: each uses only its own Pst
// slice, so the hot loop needs NO workgroup barrier.  Per-wave LDS ordering
// is guaranteed by CDNA5 in-order DS processing + an explicit s_wait_dscnt
// before the re-read (and a compiler-only fence so the pipeliner cannot hoist
// the next iteration's stores above this iteration's loads).
// ---------------------------------------------------------------------------
__global__ __launch_bounds__(128) void attn_kernel(
    const _Float16* __restrict__ Q, const _Float16* __restrict__ K,
    const _Float16* __restrict__ V, const float* __restrict__ Zinv,
    const float* __restrict__ x1, const float* __restrict__ x2,
    const float* __restrict__ Wsc, const float* __restrict__ gamma,
    float* __restrict__ out)
{
    __shared__ __align__(16) _Float16 Pst[4][16][32];   // 4 KB: P tile per wave
    __shared__ float Outs[4][16][64];                   // 16 KB: OUT staging
    __shared__ float sWsc[32*64];                       // 8 KB
    for (int i = threadIdx.x; i < 32*64; i += 128) sWsc[i] = Wsc[i];

    int lane = threadIdx.x & 31;
    int wv   = threadIdx.x >> 5;
    int w    = blockIdx.x * 4 + wv;
    int b    = w >> 8;                 // 256 n-tiles per batch
    int n0   = (w & 255) << 4;
    int hl   = lane & 15, sel = lane >> 4;

    // A operand: q rows for this n-tile (fixed across the whole m loop)
    const _Float16* qr = Q + ((size_t)b*HW + n0 + hl)*32 + sel*16;
    v8h qlo = *(const v8h*)(qr);
    v8h qhi = *(const v8h*)(qr + 8);
    v16h aq;
#pragma unroll
    for (int i = 0; i < 8; ++i) { aq[i] = qlo[i]; aq[8 + i] = qhi[i]; }

    v8f acc[4] = {v8f{}, v8f{}, v8f{}, v8f{}};
    const _Float16* kb = K + (size_t)b*HW*32;
    const _Float16* vb = V + (size_t)b*64*HW;
    const float*    zb = Zinv + (size_t)b*HW;
    __syncthreads();   // Wsc staged (only cross-wave dependency before epilogue)

    for (int m0 = 0; m0 < HW; m0 += 32) {
        // S tiles: columns [m0,m0+16) and [m0+16,m0+32)
        v16h bk0 = *(const v16h*)(kb + (size_t)(m0 + hl)*32      + sel*16);
        v16h bk1 = *(const v16h*)(kb + (size_t)(m0 + 16 + hl)*32 + sel*16);
        v8f s0 = {}, s1 = {};
        s0 = __builtin_amdgcn_wmma_f32_16x16x32_f16(false, aq, false, bk0, (short)0, s0, false, false);
        s1 = __builtin_amdgcn_wmma_f32_16x16x32_f16(false, aq, false, bk1, (short)0, s1, false, false);

        float rz0 = zb[m0 + hl];
        float rz1 = zb[m0 + 16 + hl];
#pragma unroll
        for (int r = 0; r < 8; ++r) {
            int row = sel*8 + r;
            Pst[wv][row][hl]      = (_Float16)(__expf(s0[r]) * rz0);
            Pst[wv][row][16 + hl] = (_Float16)(__expf(s1[r]) * rz1);
        }
        // Per-wave store->load visibility: explicit CDNA5 split-counter wait.
        asm volatile("s_wait_dscnt 0x0" ::: "memory");

        // Re-read P in A-operand layout
        v8h plo = *(const v8h*)(&Pst[wv][hl][sel*16]);
        v8h phi = *(const v8h*)(&Pst[wv][hl][sel*16 + 8]);
        v16h ap;
#pragma unroll
        for (int i = 0; i < 8; ++i) { ap[i] = plo[i]; ap[8 + i] = phi[i]; }
        // Compiler-only fence: next iteration's Pst stores must stay after the
        // loads above (hardware WAR is safe by in-order per-wave DS).
        asm volatile("" ::: "memory");

        // OUT accumulation: B operand = v rows (contiguous in m)
#pragma unroll
        for (int t = 0; t < 4; ++t) {
            v16h bv = *(const v16h*)(vb + (size_t)(t*16 + hl)*HW + m0 + sel*16);
            acc[t] = __builtin_amdgcn_wmma_f32_16x16x32_f16(false, ap, false, bv, (short)0, acc[t], false, false);
        }
    }

    // ---- epilogue: OUT -> LDS, residual, final projection ----
#pragma unroll
    for (int t = 0; t < 4; ++t)
#pragma unroll
        for (int r = 0; r < 8; ++r)
            Outs[wv][sel*8 + r][t*16 + hl] = acc[t][r];
    asm volatile("s_wait_dscnt 0x0" ::: "memory");

    float g = gamma[0];
    for (int cc = lane; cc < 64; cc += 32)
        for (int nl = 0; nl < 16; ++nl) {
            float xv = (cc < 32) ? x1[(size_t)(b*32 + cc)*HW + n0 + nl]
                                 : x2[(size_t)(b*32 + cc - 32)*HW + n0 + nl];
            Outs[wv][nl][cc] = g * Outs[wv][nl][cc] + xv;
        }
    asm volatile("s_wait_dscnt 0x0" ::: "memory");

    int o = lane;   // 32 output channels, one per lane
    for (int nl = 0; nl < 16; ++nl) {
        float a = 0.f;
#pragma unroll
        for (int c = 0; c < 64; ++c) a += sWsc[o*64 + c] * Outs[wv][nl][c];
        out[((size_t)b*32 + o)*HW + n0 + nl] = a;
    }
}

// ---------------------------------------------------------------------------
extern "C" void kernel_launch(void* const* d_in, const int* in_sizes, int n_in,
                              void* d_out, int out_size, void* d_ws, size_t ws_size,
                              hipStream_t stream) {
    const float* x1    = (const float*)d_in[0];
    const float* x2    = (const float*)d_in[1];
    const float* Wq    = (const float*)d_in[2];
    const float* bq    = (const float*)d_in[3];
    const float* Wk    = (const float*)d_in[4];
    const float* bk    = (const float*)d_in[5];
    const float* Wv    = (const float*)d_in[6];
    const float* Wsc   = (const float*)d_in[7];
    const float* gamma = (const float*)d_in[8];
    float* out = (float*)d_out;

    // workspace: Q (2MB f16) | K (2MB f16) | V (4MB f16) | Zinv (128KB f32)
    _Float16* Q = (_Float16*)d_ws;
    _Float16* K = Q + (size_t)BATCH * HW * 32;
    _Float16* V = K + (size_t)BATCH * HW * 32;
    float* Zinv = (float*)(V + (size_t)BATCH * 64 * HW);

    proj_kernel<<<BATCH * HW / 256, 256, 0, stream>>>(x1, x2, Wq, bq, Wk, bk, Wv, Q, K, V);
    colstats_kernel<<<BATCH * (HW / 16) / 4, 128, 0, stream>>>(Q, K, Zinv);
    attn_kernel<<<BATCH * (HW / 16) / 4, 128, 0, stream>>>(Q, K, V, Zinv, x1, x2, Wsc, gamma, out);
}
